// SelfAttentionV3_48627619725380
// MI455X (gfx1250) — matrix operand order repaired and checked by reading, not compiled, not run on hardware
//
#include <hip/hip_runtime.h>

// ---------------------------------------------------------------------------
// Self-attention for MI455X (gfx1250): B=4, S=2048, H=1024, single head.
// All matmuls via v_wmma_f32_16x16x32_bf16 (bf16 operands, f32 accumulate).
// GEMM: 128x256 block tile, 8 waves (2x4), 64x64 per wave (16 WMMA / K-step),
// double-buffered LDS (one barrier per K-step), register global prefetch.
// ---------------------------------------------------------------------------

typedef __attribute__((ext_vector_type(16))) __bf16 bf16x16;
typedef __attribute__((ext_vector_type(8)))  __bf16 bf16x8;
typedef __attribute__((ext_vector_type(4)))  __bf16 bf16x4;
typedef __attribute__((ext_vector_type(8)))  float  floatx8;

#define HID 1024
#define SEQ 2048
#define BATCH 4

// ---------------------------------------------------------------------------
// f32 -> bf16 elementwise convert (4 elements / thread, vectorized)
// ---------------------------------------------------------------------------
__global__ __launch_bounds__(256) void cvt_f32_bf16(const float* __restrict__ x,
                                                    __bf16* __restrict__ y) {
  long long i = ((long long)blockIdx.x * 256 + threadIdx.x) * 4;
  float4 f = *(const float4*)(x + i);
  bf16x4 o;
  o[0] = (__bf16)f.x; o[1] = (__bf16)f.y; o[2] = (__bf16)f.z; o[3] = (__bf16)f.w;
  *(bf16x4*)(y + i) = o;
}

// ---------------------------------------------------------------------------
// f32 -> bf16 transposed convert (D x D), 32x32 LDS tile
// ---------------------------------------------------------------------------
__global__ __launch_bounds__(256) void transpose_cvt_bf16(const float* __restrict__ W,
                                                          __bf16* __restrict__ WT,
                                                          int D) {
  __shared__ float t[32][33];
  const int bx = blockIdx.x * 32, by = blockIdx.y * 32;
  const int tx = threadIdx.x & 31, ty = threadIdx.x >> 5;  // 32x8
#pragma unroll
  for (int i = 0; i < 32; i += 8)
    t[ty + i][tx] = W[(long long)(by + ty + i) * D + bx + tx];
  __syncthreads();
#pragma unroll
  for (int i = 0; i < 32; i += 8)
    WT[(long long)(bx + ty + i) * D + by + tx] = (__bf16)t[tx][ty + i];
}

// ---------------------------------------------------------------------------
// LDS fragment loaders (padded row stride = 40 bf16 -> conflict-free b128s)
// A 16x32 bf16 frag: lanes 0-15 hold K 0..7 & 16..23, lanes 16-31 K 8..15 & 24..31
// B 32x16 bf16 frag: lanes 0-15 hold K 0..7 & 16..23, lanes 16-31 K 8..15 & 24..31 (N = lane%16)
// ---------------------------------------------------------------------------
__device__ __forceinline__ bf16x16 ldsA_frag(const __bf16* As, int mrow, int lane) {
  const int k0 = (lane & 16) ? 8 : 0;
  bf16x8 lo = *(const bf16x8*)(As + mrow * 40 + k0);
  bf16x8 hi = *(const bf16x8*)(As + mrow * 40 + k0 + 16);
  return __builtin_shufflevector(lo, hi, 0, 1, 2, 3, 4, 5, 6, 7, 8, 9, 10, 11, 12, 13, 14, 15);
}
__device__ __forceinline__ bf16x16 ldsB_frag(const __bf16* Bs, int nrow, int lane) {
  const int k0 = (lane & 16) ? 8 : 0;
  bf16x8 lo = *(const bf16x8*)(Bs + nrow * 40 + k0);
  bf16x8 hi = *(const bf16x8*)(Bs + nrow * 40 + k0 + 16);
  return __builtin_shufflevector(lo, hi, 0, 1, 2, 3, 4, 5, 6, 7, 8, 9, 10, 11, 12, 13, 14, 15);
}

// ---------------------------------------------------------------------------
// Generic WMMA GEMM:  C[z] = A[z](MxK, row-major) * Bt[z](NxK, row-major)^T
// Block tile 128x256, BK=32 (one WMMA K-step). 8 waves in 2(M) x 4(N); each
// wave owns a 64x64 C tile = 4x4 16x16 f32 accumulators (16 WMMAs / K-step).
// Double-buffered LDS -> single barrier per K-step; next tile prefetched to
// registers two steps ahead so global loads overlap the WMMA clause.
// Epilogues: bf16 / bf16-transposed / f32, optional f32 bias over N.
// All problem dims are multiples of tile sizes (no bounds checks needed).
// ---------------------------------------------------------------------------
template <bool TRANS_OUT, bool F32_OUT>
__global__ __launch_bounds__(256) void gemm_bf16_wmma(
    const __bf16* __restrict__ A, const __bf16* __restrict__ Bt,
    void* __restrict__ Cv, const float* __restrict__ bias,
    int K, int lda, int ldb, int ldc,
    long long sA, long long sB, long long sC) {
  __shared__ __bf16 As[2][128 * 40];
  __shared__ __bf16 Bs[2][256 * 40];

  const int tid = threadIdx.x;
  const int lane = tid & 31;
  const int wave = tid >> 5;
  const int waveM = wave & 1;   // 0..1  (64 rows each)
  const int waveN = wave >> 1;  // 0..3  (64 cols each)
  const int z = blockIdx.z;

  const __bf16* Ab = A + (long long)z * sA + (long long)(blockIdx.y * 128) * lda;
  const __bf16* Bb = Bt + (long long)z * sB + (long long)(blockIdx.x * 256) * ldb;

  const int r0 = tid >> 2;        // staging row 0..63 (+64k for further rows)
  const int c0 = (tid & 3) << 3;  // col chunk 0,8,16,24

  const int KT = K >> 5;

  bf16x8 pa[2], pb[4];
  // prologue: tile 0 -> regs -> LDS buf 0; tile 1 -> regs
  {
    const int ko = c0;
    pa[0] = *(const bf16x8*)(Ab + (long long)r0 * lda + ko);
    pa[1] = *(const bf16x8*)(Ab + (long long)(r0 + 64) * lda + ko);
#pragma unroll
    for (int k = 0; k < 4; ++k)
      pb[k] = *(const bf16x8*)(Bb + (long long)(r0 + 64 * k) * ldb + ko);
  }
  *(bf16x8*)(&As[0][r0 * 40 + c0]) = pa[0];
  *(bf16x8*)(&As[0][(r0 + 64) * 40 + c0]) = pa[1];
#pragma unroll
  for (int k = 0; k < 4; ++k)
    *(bf16x8*)(&Bs[0][(r0 + 64 * k) * 40 + c0]) = pb[k];
  if (KT > 1) {
    const int ko = 32 + c0;
    pa[0] = *(const bf16x8*)(Ab + (long long)r0 * lda + ko);
    pa[1] = *(const bf16x8*)(Ab + (long long)(r0 + 64) * lda + ko);
#pragma unroll
    for (int k = 0; k < 4; ++k)
      pb[k] = *(const bf16x8*)(Bb + (long long)(r0 + 64 * k) * ldb + ko);
  }
  __syncthreads();

  floatx8 acc[4][4] = {};

  for (int kt = 0; kt < KT; ++kt) {
    const int cur = kt & 1;
    const int nxt = cur ^ 1;

    if (kt + 1 < KT) {
      // stage tile kt+1 (already in regs) into the other LDS buffer
      *(bf16x8*)(&As[nxt][r0 * 40 + c0]) = pa[0];
      *(bf16x8*)(&As[nxt][(r0 + 64) * 40 + c0]) = pa[1];
#pragma unroll
      for (int k = 0; k < 4; ++k)
        *(bf16x8*)(&Bs[nxt][(r0 + 64 * k) * 40 + c0]) = pb[k];
      if (kt + 2 < KT) {  // prefetch tile kt+2 into regs
        const int ko = (kt + 2) * 32 + c0;
        pa[0] = *(const bf16x8*)(Ab + (long long)r0 * lda + ko);
        pa[1] = *(const bf16x8*)(Ab + (long long)(r0 + 64) * lda + ko);
#pragma unroll
        for (int k = 0; k < 4; ++k)
          pb[k] = *(const bf16x8*)(Bb + (long long)(r0 + 64 * k) * ldb + ko);
      }
    }

    bf16x16 afr[4];
#pragma unroll
    for (int i = 0; i < 4; ++i)
      afr[i] = ldsA_frag(As[cur], waveM * 64 + i * 16 + (lane & 15), lane);
#pragma unroll
    for (int j = 0; j < 4; ++j) {
      bf16x16 bfr = ldsB_frag(Bs[cur], waveN * 64 + j * 16 + (lane & 15), lane);
#pragma unroll
      for (int i = 0; i < 4; ++i)
        acc[i][j] = __builtin_amdgcn_wmma_f32_16x16x32_bf16(
            false, afr[i], false, bfr, (short)0, acc[i][j], false, false);
    }

    __syncthreads();  // single barrier per K-step (double-buffered LDS)
  }

  // Epilogue. C/D layout: VGPR r -> M=r (lanes 0-15) / M=r+8 (lanes 16-31), N=lane%16.
  const long long cz = (long long)z * sC;
  const int mW = blockIdx.y * 128 + waveM * 64;
  const int nW = blockIdx.x * 256 + waveN * 64;
  const int mOff = (lane & 16) ? 8 : 0;
  const int nLane = lane & 15;

#pragma unroll
  for (int i = 0; i < 4; ++i) {
#pragma unroll
    for (int j = 0; j < 4; ++j) {
      const int n = nW + j * 16 + nLane;
      const float bb = bias ? bias[n] : 0.0f;
      if (TRANS_OUT) {
        // per-lane contiguous 8 M-values -> single 16B store of C^T row n
        bf16x8 pk;
#pragma unroll
        for (int r = 0; r < 8; ++r) pk[r] = (__bf16)(acc[i][j][r] + bb);
        const int m0 = mW + i * 16 + mOff;
        *(bf16x8*)((__bf16*)Cv + cz + (long long)n * ldc + m0) = pk;
      } else if (F32_OUT) {
#pragma unroll
        for (int r = 0; r < 8; ++r) {
          const int m = mW + i * 16 + mOff + r;
          ((float*)Cv)[cz + (long long)m * ldc + n] = acc[i][j][r] + bb;
        }
      } else {
#pragma unroll
        for (int r = 0; r < 8; ++r) {
          const int m = mW + i * 16 + mOff + r;
          ((__bf16*)Cv)[cz + (long long)m * ldc + n] = (__bf16)(acc[i][j][r] + bb);
        }
      }
    }
  }
}

// ---------------------------------------------------------------------------
// Masked, scaled softmax over rows of length SEQ. One block (256 thr) per row;
// each thread owns 8 contiguous elements. s = mask ? 1e-20 : raw/32.
// ---------------------------------------------------------------------------
__global__ __launch_bounds__(256) void softmax_mask_bf16(
    const float* __restrict__ scores, const unsigned char* __restrict__ mask,
    __bf16* __restrict__ attn) {
  const int row = blockIdx.x;  // query index
  const int b = blockIdx.y;    // batch
  const long long base = ((long long)b * SEQ + row) * (long long)SEQ;
  const int tid = threadIdx.x;
  const float scale = 0.03125f;  // 1/sqrt(1024)

  const float4* sp = (const float4*)(scores + base);
  float4 s0 = sp[tid * 2];
  float4 s1 = sp[tid * 2 + 1];
  const unsigned int* mp = (const unsigned int*)(mask + base);
  unsigned int mm0 = mp[tid * 2], mm1 = mp[tid * 2 + 1];

  float v[8] = {s0.x, s0.y, s0.z, s0.w, s1.x, s1.y, s1.z, s1.w};
#pragma unroll
  for (int j = 0; j < 8; ++j) {
    unsigned int mb = ((j < 4 ? (mm0 >> (8 * j)) : (mm1 >> (8 * (j - 4)))) & 0xffu);
    v[j] = mb ? 1e-20f : v[j] * scale;
  }

  __shared__ float red[256];
  float mx = v[0];
#pragma unroll
  for (int j = 1; j < 8; ++j) mx = fmaxf(mx, v[j]);
  red[tid] = mx;
  __syncthreads();
  for (int s = 128; s > 0; s >>= 1) {
    if (tid < s) red[tid] = fmaxf(red[tid], red[tid + s]);
    __syncthreads();
  }
  const float rowmax = red[0];
  __syncthreads();

  float sum = 0.0f;
#pragma unroll
  for (int j = 0; j < 8; ++j) {
    v[j] = __expf(v[j] - rowmax);
    sum += v[j];
  }
  red[tid] = sum;
  __syncthreads();
  for (int s = 128; s > 0; s >>= 1) {
    if (tid < s) red[tid] += red[tid + s];
    __syncthreads();
  }
  const float inv = 1.0f / red[0];

  bf16x8 pk;
#pragma unroll
  for (int j = 0; j < 8; ++j) pk[j] = (__bf16)(v[j] * inv);
  *(bf16x8*)(attn + base + (long long)tid * 8) = pk;
}

// ---------------------------------------------------------------------------
// Host-side orchestration
// ---------------------------------------------------------------------------
extern "C" void kernel_launch(void* const* d_in, const int* in_sizes, int n_in,
                              void* d_out, int out_size, void* d_ws, size_t ws_size,
                              hipStream_t stream) {
  const float* X = (const float*)d_in[0];
  const unsigned char* mask = (const unsigned char*)d_in[1];  // jax bool -> 1 byte
  const float* Wq = (const float*)d_in[2];
  const float* bq = (const float*)d_in[3];
  const float* Wk = (const float*)d_in[4];
  const float* bk = (const float*)d_in[5];
  const float* Wv = (const float*)d_in[6];
  const float* bv = (const float*)d_in[7];
  const float* Wo = (const float*)d_in[8];
  const float* bo = (const float*)d_in[9];
  float* out = (float*)d_out;

  const long long T = (long long)BATCH * SEQ;  // 8192 tokens
  char* w = (char*)d_ws;
  __bf16* Xbf = (__bf16*)w;            w += T * HID * 2;                      // 16 MB
  __bf16* WqT = (__bf16*)w;            w += (size_t)HID * HID * 2;            // 2 MB
  __bf16* WkT = (__bf16*)w;            w += (size_t)HID * HID * 2;
  __bf16* WvT = (__bf16*)w;            w += (size_t)HID * HID * 2;
  __bf16* WoT = (__bf16*)w;            w += (size_t)HID * HID * 2;
  __bf16* Qbf = (__bf16*)w;            w += T * HID * 2;                      // 16 MB
  __bf16* Kbf = (__bf16*)w;            w += T * HID * 2;                      // 16 MB
  __bf16* VT  = (__bf16*)w;            w += T * HID * 2;                      // 16 MB (per-batch [H][S])
  __bf16* ctx = (__bf16*)w;            w += T * HID * 2;                      // 16 MB
  float*  scoresF = (float*)w;         w += (size_t)BATCH * SEQ * SEQ * 4;    // 64 MB
  __bf16* attnbf = (__bf16*)w;         w += (size_t)BATCH * SEQ * SEQ * 2;    // 32 MB

  // 1) Convert X to bf16; convert+transpose weights to bf16 (Bt = W^T, NxK)
  cvt_f32_bf16<<<(T * HID) / 1024, 256, 0, stream>>>(X, Xbf);
  transpose_cvt_bf16<<<dim3(32, 32), 256, 0, stream>>>(Wq, WqT, HID);
  transpose_cvt_bf16<<<dim3(32, 32), 256, 0, stream>>>(Wk, WkT, HID);
  transpose_cvt_bf16<<<dim3(32, 32), 256, 0, stream>>>(Wv, WvT, HID);
  transpose_cvt_bf16<<<dim3(32, 32), 256, 0, stream>>>(Wo, WoT, HID);

  // 2) Q = X*Wq + bq  (8192x1024, bf16 out)
  gemm_bf16_wmma<false, false><<<dim3(HID / 256, T / 128, 1), 256, 0, stream>>>(
      Xbf, WqT, Qbf, bq, HID, HID, HID, HID, 0, 0, 0);
  // 3) K = X*Wk + bk
  gemm_bf16_wmma<false, false><<<dim3(HID / 256, T / 128, 1), 256, 0, stream>>>(
      Xbf, WkT, Kbf, bk, HID, HID, HID, HID, 0, 0, 0);
  // 4) V^T = (X*Wv + bv)^T, stored per batch as [HID][SEQ] (batched, transposed store)
  gemm_bf16_wmma<true, false><<<dim3(HID / 256, SEQ / 128, BATCH), 256, 0, stream>>>(
      Xbf, WvT, VT, bv, HID, HID, HID, /*ldcT=*/SEQ,
      (long long)SEQ * HID, 0, (long long)HID * SEQ);

  // 5) scores = Q * K^T (K rows used directly as Bt), f32 out, per batch
  gemm_bf16_wmma<false, true><<<dim3(SEQ / 256, SEQ / 128, BATCH), 256, 0, stream>>>(
      Qbf, Kbf, scoresF, nullptr, HID, HID, HID, SEQ,
      (long long)SEQ * HID, (long long)SEQ * HID, (long long)SEQ * SEQ);

  // 6) attn = softmax(where(mask, 1e-20, scores/32)), bf16 out
  softmax_mask_bf16<<<dim3(SEQ, BATCH), 256, 0, stream>>>(scoresF, mask, attnbf);

  // 7) ctx = attn * V  (Bt = V^T, direct), bf16 out, per batch
  gemm_bf16_wmma<false, false><<<dim3(HID / 256, SEQ / 128, BATCH), 256, 0, stream>>>(
      attnbf, VT, ctx, nullptr, SEQ, SEQ, SEQ, HID,
      (long long)SEQ * SEQ, (long long)HID * SEQ, (long long)SEQ * HID);

  // 8) out = ctx * Wo + bo, f32 out (flat over all tokens)
  gemm_bf16_wmma<false, true><<<dim3(HID / 256, T / 128, 1), 256, 0, stream>>>(
      ctx, WoT, out, bo, HID, HID, HID, HID, 0, 0, 0);
}
